// fakeSuctionFCGQCNN_33543694582172
// MI455X (gfx1250) — matrix operand order repaired and checked by reading, not compiled, not run on hardware
//
#include <hip/hip_runtime.h>
#include <hip/hip_bf16.h>

typedef __attribute__((ext_vector_type(16))) _Float16 v16h;
typedef __attribute__((ext_vector_type(8)))  _Float16 v8h;
typedef __attribute__((ext_vector_type(8)))  float    v8f;

// ws layout (bytes)
#define WS_F    0u                    // features: 9216 x 1024 f16
#define WS_W1L  18874368u             // conv1 B: [16][160] f16 (ky*16+kx, zero padded)
#define WS_W2L  18879488u             // conv2 B: [16][416] f16 (tap*16+ci, tap25 = zeros)
#define WS_W3T  18892800u             // fc1 transposed: [128][1024] f16
#define WS_POSE 19154944u             // float: pose branch + b_fc2 folded

__device__ __forceinline__ v16h join16(const v8h lo, const v8h hi) {
    v16h r;
#pragma unroll
    for (int j = 0; j < 8; ++j) { r[j] = lo[j]; r[8 + j] = hi[j]; }
    return r;
}

// ---------------- prep: weight conversion + pose scalar ----------------
__global__ void prep_kernel(const float* __restrict__ w1, const float* __restrict__ w2,
                            const float* __restrict__ wfc1, const float* __restrict__ zin,
                            const float* __restrict__ wpose, const float* __restrict__ bpose,
                            const float* __restrict__ wfc2, const float* __restrict__ bfc2,
                            char* __restrict__ ws) {
    int gid = blockIdx.x * blockDim.x + threadIdx.x;
    _Float16* W1L = (_Float16*)(ws + WS_W1L);
    _Float16* W2L = (_Float16*)(ws + WS_W2L);
    _Float16* W3T = (_Float16*)(ws + WS_W3T);
    if (gid < 2560) {                                   // conv1: [16][160]
        int co = gid / 160, k = gid % 160, ky = k >> 4, kx = k & 15;
        float v = (ky < 9 && kx < 9) ? w1[co * 81 + ky * 9 + kx] : 0.f;
        W1L[gid] = (_Float16)v;
    } else if (gid < 2560 + 6656) {                     // conv2: [16][416]
        int i = gid - 2560;
        int co = i / 416, k = i % 416, tap = k >> 4, ci = k & 15;
        float v = (tap < 25) ? w2[co * 400 + ci * 25 + tap] : 0.f;
        W2L[i] = (_Float16)v;
    } else if (gid < 2560 + 6656 + 131072) {            // fc1 transpose: [128][1024]
        int i = gid - (2560 + 6656);
        int n = i >> 10, k = i & 1023;
        W3T[i] = (_Float16)wfc1[k * 128 + n];
    } else if (gid == 2560 + 6656 + 131072) {           // pose scalar (z identical for all patches)
        float z0 = zin[0], s = bfc2[0];
        for (int j = 0; j < 16; ++j) {
            float pj = z0 * wpose[j] + bpose[j];
            if (pj < 0.f) pj = 0.f;
            s += pj * wfc2[128 + j];
        }
        *(float*)(ws + WS_POSE) = s;
    }
}

__global__ void zero_kernel(float* __restrict__ out) {
    out[blockIdx.x * 256 + threadIdx.x] = 0.f;
}

// ---------------- conv chain: one workgroup per patch ----------------
__global__ __launch_bounds__(256) void conv_kernel(const float* __restrict__ x,
                                                   const float* __restrict__ bconv1,
                                                   const float* __restrict__ bconv2,
                                                   char* __restrict__ ws) {
    __shared__ __align__(16) _Float16 sP[41 * 48];        // patch + 4px zero halo
    __shared__ __align__(16) _Float16 sB1[16 * 160];      // conv1 weights [n][k]
    __shared__ __align__(16) _Float16 sB2[16 * 416];      // conv2 weights [n][k]
    __shared__ __align__(16) _Float16 sO1[512 * 16];      // conv1 half-output (reused as conv2 out)
    __shared__ __align__(16) _Float16 sI2[20 * 20 * 16];  // pooled conv1 + 2px zero halo, ch innermost

    const int tid  = threadIdx.x;
    const int lane = tid & 31;
    const int wv   = tid >> 5;          // wave 0..7
    const int hf   = lane >> 4;         // lane half (selects K sub-range / row block)
    const int mrow = lane & 15;         // A row within tile == B column (channel)
    const int p    = blockIdx.x;
    const int py   = p / 96, px = p % 96;

    // ---- phase 0: zero halos, load patch (f32->f16), load weights into LDS
    for (int i = tid; i < 41 * 48 / 2; i += 256) ((unsigned int*)sP)[i] = 0u;
    for (int i = tid; i < 20 * 20 * 16 / 2; i += 256) ((unsigned int*)sI2)[i] = 0u;
    {
        const unsigned int* g1 = (const unsigned int*)(ws + WS_W1L);
        for (int i = tid; i < 16 * 160 / 2; i += 256) ((unsigned int*)sB1)[i] = g1[i];
        const unsigned int* g2 = (const unsigned int*)(ws + WS_W2L);
        for (int i = tid; i < 16 * 416 / 2; i += 256) ((unsigned int*)sB2)[i] = g2[i];
    }
    __syncthreads();
    for (int i = tid; i < 1024; i += 256) {
        int r = i >> 5, c = i & 31;
        sP[(r + 4) * 48 + (c + 4)] = (_Float16)x[(py + r) * 128 + (px + c)];
    }
    __syncthreads();

    const float bc1 = bconv1[mrow];
    const float bc2 = bconv2[mrow];

    // ---- conv1 (M=1024,N=16,K=160 = 5 k-steps of {2 ky taps x 16 kx}) + pool, in 2 half passes
    for (int ph = 0; ph < 2; ++ph) {
        for (int mi = 0; mi < 4; ++mi) {
            int mt = ph * 32 + wv * 4 + mi;                    // M-tile
            int y  = mt >> 1;
            int xx = ((mt & 1) << 4) + mrow;                   // output col for this lane's A row
            v8f acc = {};
#pragma unroll
            for (int s = 0; s < 5; ++s) {
                // A: elems 0..7 -> ky=2s, kx=hf*8+j ; elems 8..15 -> ky=2s+1
                const _Float16* p0 = &sP[(y + 2 * s) * 48 + xx + hf * 8];
                const _Float16* p1 = p0 + 48;
                v16h a;
#pragma unroll
                for (int j = 0; j < 8; ++j) { a[j] = p0[j]; a[8 + j] = p1[j]; }
                // B: lane col = mrow, k = s*32 + hf*8 + e (e<8), s*32 + 16 + hf*8 + (e-8) (e>=8)
                const v8h* bp = (const v8h*)&sB1[mrow * 160 + s * 32 + hf * 8];
                v16h b = join16(bp[0], bp[2]);
                acc = __builtin_amdgcn_wmma_f32_16x16x32_f16(false, a, false, b,
                                                             (short)0, acc, false, false);
            }
#pragma unroll
            for (int r = 0; r < 8; ++r) {                      // D: row = mt*16 + r + hf*8, col = mrow
                int Mg = mt * 16 + r + hf * 8;
                float v = acc[r] + bc1;
                if (v < 0.f) v = 0.f;
                sO1[(Mg - ph * 512) * 16 + mrow] = (_Float16)v;
            }
        }
        __syncthreads();
        // 2x2 maxpool of this half into sI2 (with +2 halo offset)
        for (int i = tid; i < 2048; i += 256) {
            int ch = i & 15, pos = i >> 4;
            int yy = (pos >> 4) + ph * 8, xc = pos & 15;
            int b0 = ((yy * 2 - ph * 16) * 32 + xc * 2) * 16 + ch;
            _Float16 m0 = sO1[b0],       m1 = sO1[b0 + 16];
            _Float16 m2 = sO1[b0 + 512], m3 = sO1[b0 + 528];
            _Float16 m = m0 > m1 ? m0 : m1;
            if (m2 > m) m = m2;
            if (m3 > m) m = m3;
            sI2[((yy + 2) * 20 + (xc + 2)) * 16 + ch] = m;
        }
        __syncthreads();
    }

    // ---- conv2 (M=256,N=16,K=416 = 13 k-steps of {2 taps x 16 ch})
    _Float16* sO2 = sO1;                                       // 16*16*16 fits in sO1
    for (int mi = 0; mi < 2; ++mi) {
        int mt = wv * 2 + mi;                                  // == output row y
        int xx = mrow;                                         // output col
        v8f acc = {};
#pragma unroll
        for (int s = 0; s < 13; ++s) {
            int t0 = 2 * s, t1 = 2 * s + 1;
            if (t1 > 24) t1 = 24;                              // pad tap -> B is zero there
            int ky0 = t0 / 5, kx0 = t0 % 5, ky1 = t1 / 5, kx1 = t1 % 5;
            const v8h* a0 = (const v8h*)&sI2[((mt + ky0) * 20 + (xx + kx0)) * 16 + hf * 8];
            const v8h* a1 = (const v8h*)&sI2[((mt + ky1) * 20 + (xx + kx1)) * 16 + hf * 8];
            v16h a = join16(*a0, *a1);
            const v8h* bp = (const v8h*)&sB2[mrow * 416 + s * 32 + hf * 8];
            v16h b = join16(bp[0], bp[2]);
            acc = __builtin_amdgcn_wmma_f32_16x16x32_f16(false, a, false, b,
                                                         (short)0, acc, false, false);
        }
#pragma unroll
        for (int r = 0; r < 8; ++r) {                          // row y=mt, col x=r+hf*8, ch=mrow
            float v = acc[r] + bc2;
            if (v < 0.f) v = 0.f;
            sO2[(mt * 16 + r + hf * 8) * 16 + mrow] = (_Float16)v;
        }
    }
    __syncthreads();

    // ---- pool2 + write flat features (C,H,W order) to ws
    _Float16* F = (_Float16*)(ws + WS_F);
    for (int i = tid; i < 1024; i += 256) {
        int ch = i >> 6, rem = i & 63, yy = rem >> 3, xc = rem & 7;
        int b0 = ((yy * 2) * 16 + xc * 2) * 16 + ch;
        _Float16 m0 = sO2[b0],       m1 = sO2[b0 + 16];
        _Float16 m2 = sO2[b0 + 256], m3 = sO2[b0 + 272];
        _Float16 m = m0 > m1 ? m0 : m1;
        if (m2 > m) m = m2;
        if (m3 > m) m = m3;
        F[(unsigned)p * 1024u + (unsigned)i] = m;              // i == ch*64 + yy*8 + xc
    }
}

// ---------------- fc1 + fc2 fused: 16 patches per workgroup ----------------
__global__ __launch_bounds__(128) void fc_kernel(const float* __restrict__ bfc1,
                                                 const float* __restrict__ wfc2,
                                                 const char* __restrict__ ws,
                                                 float* __restrict__ out) {
    __shared__ float qs[16];
    const int tid  = threadIdx.x;
    const int lane = tid & 31;
    const int wv   = tid >> 5;          // 4 waves, 2 N-tiles each -> 128 fc1 outputs
    const int hf   = lane >> 4;
    const int mrow = lane & 15;
    const int wg   = blockIdx.x;

    const _Float16* F   = (const _Float16*)(ws + WS_F);
    const _Float16* W3T = (const _Float16*)(ws + WS_W3T);
    const float pose = *(const float*)(ws + WS_POSE);
    if (tid < 16) qs[tid] = pose;
    __syncthreads();

    const unsigned row = (unsigned)(wg * 16 + mrow);
    const _Float16* Arow = F + row * 1024u;
    const int n0 = (wv * 2) * 16 + mrow, n1 = (wv * 2 + 1) * 16 + mrow;
    const _Float16* B0 = W3T + (unsigned)n0 * 1024u;
    const _Float16* B1 = W3T + (unsigned)n1 * 1024u;

    v8f acc0 = {}, acc1 = {};
    for (int s = 0; s < 32; ++s) {
        const v8h* ap = (const v8h*)(Arow + s * 32);
        v16h a = join16(ap[2 * hf], ap[2 * hf + 1]);           // k = hf*16..
        const v8h* bp0 = (const v8h*)(B0 + s * 32 + hf * 16);
        v16h b0 = join16(bp0[0], bp0[1]);
        const v8h* bp1 = (const v8h*)(B1 + s * 32 + hf * 16);
        v16h b1 = join16(bp1[0], bp1[1]);
        acc0 = __builtin_amdgcn_wmma_f32_16x16x32_f16(false, a, false, b0, (short)0, acc0, false, false);
        acc1 = __builtin_amdgcn_wmma_f32_16x16x32_f16(false, a, false, b1, (short)0, acc1, false, false);
    }

    const float bias0 = bfc1[n0], bias1 = bfc1[n1];
    const float w0 = wfc2[n0], w1 = wfc2[n1];
    float val[8];
#pragma unroll
    for (int r = 0; r < 8; ++r) {
        float g0 = acc0[r] + bias0; if (g0 < 0.f) g0 = 0.f;
        float g1 = acc1[r] + bias1; if (g1 < 0.f) g1 = 0.f;
        val[r] = g0 * w0 + g1 * w1;                            // patch m = r + hf*8
    }
#pragma unroll
    for (int off = 1; off < 16; off <<= 1)
#pragma unroll
        for (int r = 0; r < 8; ++r) val[r] += __shfl_xor(val[r], off, 32);
    if (mrow == 0) {
#pragma unroll
        for (int r = 0; r < 8; ++r) atomicAdd(&qs[r + hf * 8], val[r]);
    }
    __syncthreads();
    if (tid < 16) {
        int pp = wg * 16 + tid;
        int pi = pp / 96, pj = pp % 96;
        out[(pi + 16) * 128 + (pj + 16)] = qs[tid];
    }
}

extern "C" void kernel_launch(void* const* d_in, const int* in_sizes, int n_in,
                              void* d_out, int out_size, void* d_ws, size_t ws_size,
                              hipStream_t stream) {
    const float* x     = (const float*)d_in[0];
    const float* z     = (const float*)d_in[1];
    const float* w1    = (const float*)d_in[2];
    const float* b1    = (const float*)d_in[3];
    const float* w2    = (const float*)d_in[4];
    const float* b2    = (const float*)d_in[5];
    const float* wfc1  = (const float*)d_in[6];
    const float* bfc1  = (const float*)d_in[7];
    const float* wpose = (const float*)d_in[8];
    const float* bpose = (const float*)d_in[9];
    const float* wfc2  = (const float*)d_in[10];
    const float* bfc2  = (const float*)d_in[11];
    float* out = (float*)d_out;
    char*  ws  = (char*)d_ws;

    prep_kernel<<<(2560 + 6656 + 131072 + 1 + 255) / 256, 256, 0, stream>>>(
        w1, w2, wfc1, z, wpose, bpose, wfc2, bfc2, ws);
    zero_kernel<<<64, 256, 0, stream>>>(out);
    conv_kernel<<<9216, 256, 0, stream>>>(x, b1, b2, ws);
    fc_kernel<<<576, 128, 0, stream>>>(bfc1, wfc2, ws, out);
}
